// BiLSTM_CRF_64278480552025
// MI455X (gfx1250) — matrix operand order repaired and checked by reading, not compiled, not run on hardware
//
#include <hip/hip_runtime.h>
#include <hip/hip_bf16.h>
#include <stdint.h>

// ---------------- problem constants ----------------
#define T_LEN   16384
#define D1      450     // sentence1 width
#define D1P     480     // padded to mult of 32
#define DENSE_N 200
#define DENSE_NP 208    // padded to mult of 16
#define D2      415     // sentence2 width
#define DIN     615
#define DINP    640     // padded x width (mult of 32)
#define HID     512
#define HH      256
#define G4      1024    // 4*HH gates
#define KTAGS   11
#define START_TAG 9
#define STOP_TAG  10
#define NEGV    (-10000.0f)

// ---------------- WMMA fragment types (gfx1250, wave32) ----------------
typedef __attribute__((ext_vector_type(16))) __bf16 v16bf;
typedef __attribute__((ext_vector_type(8)))  float  v8f;
// TDM descriptor group types (probe: ROCm7.2 = 5-arg builtin, clang-23 = 6-arg)
typedef __attribute__((ext_vector_type(4))) unsigned int v4u;
typedef __attribute__((ext_vector_type(8))) int          v8i;
typedef __attribute__((ext_vector_type(4))) int          v4i;

union FragAB { v16bf v; uint4 q[2]; };
union FragC  { v8f   v; float f[8]; };

__device__ __forceinline__ unsigned short f2bf(float f) {
  unsigned int u = __float_as_uint(f);
  u += 0x7FFFu + ((u >> 16) & 1u);          // round-to-nearest-even
  return (unsigned short)(u >> 16);
}
__device__ __forceinline__ float bf2f(unsigned short h) {
  return __uint_as_float(((unsigned int)h) << 16);
}
__device__ __forceinline__ float sigmoidf(float x) {
  return 1.0f / (1.0f + __expf(-x));
}

// ---------------- convert fp32 -> bf16 with zero padding ----------------
__global__ void cvt_pad_bf16(const float* __restrict__ src, unsigned short* __restrict__ dst,
                             int Rs, int Cs, int Rd, int Cd) {
  int idx = blockIdx.x * blockDim.x + threadIdx.x;
  if (idx >= Rd * Cd) return;
  int r = idx / Cd, c = idx - r * Cd;
  float v = (r < Rs && c < Cs) ? src[(size_t)r * Cs + c] : 0.0f;
  dst[idx] = f2bf(v);
}

// ---------------- fill Xb cols [200,640): sentence2 + zero pad ----------------
__global__ void fill_x_kernel(const float* __restrict__ s2, unsigned short* __restrict__ Xb) {
  int idx = blockIdx.x * blockDim.x + threadIdx.x;
  const int COLS = DINP - DENSE_N; // 440
  if (idx >= T_LEN * COLS) return;
  int r = idx / COLS, c = idx - r * COLS;
  int col = DENSE_N + c;
  float v = (col < DIN) ? s2[(size_t)r * D2 + (col - DENSE_N)] : 0.0f;
  Xb[(size_t)r * DINP + col] = f2bf(v);
}

__global__ void add_bias_kernel(const float* __restrict__ a, const float* __restrict__ b,
                                float* __restrict__ o, int n) {
  int i = blockIdx.x * blockDim.x + threadIdx.x;
  if (i < n) o[i] = a[i] + b[i];
}

// ---------------- generic WMMA bf16 GEMM: out[m,n] = sum_k A[m,k]*B[n,k] + bias[n] -------
// A: M x Kp bf16 row-major (M mult of 16, Kp mult of 32, Kp <= DINP)
// B: Np x Kp bf16 row-major (weight rows = output columns, Np mult of 16)
// block = 256 threads (8 waves); wave w -> 16x16 tile at (blockIdx.x*16, blockIdx.y*128 + w*16)
// The block's 16 x Kp A-tile is DMA'd into LDS once by the Tensor Data Mover
// (TENSOR_LOAD_TO_LDS, tracked with TENSORcnt); all 8 waves then read A fragments
// from LDS (ds_load_b128) instead of 8x-redundant global loads. B streams from
// global (its reuse is across M-blocks; the whole weight set is L2-resident).
__global__ __launch_bounds__(256)
void wmma_gemm_kernel(const unsigned short* __restrict__ A,
                      const unsigned short* __restrict__ B,
                      const float* __restrict__ bias,
                      float* __restrict__ outF, int ldf,
                      unsigned short* __restrict__ outB, int ldb, int col_off,
                      int M, int Nreal, int Np, int Kp, int relu) {
  __shared__ unsigned short As[16 * DINP];

  const int lane = threadIdx.x & 31;
  const int wave = threadIdx.x >> 5;
  const int m0 = blockIdx.x * 16;
  const int n0 = blockIdx.y * 128 + wave * 16;

  // LDS byte offset of As: flat LDS pointers carry the wave-relative LDS offset
  // in the low 32 bits (the high half selects the aperture).
  // The ptrtoint also captures As so alias analysis keeps the ds_loads below live.
  const unsigned lds_off = (unsigned)(uintptr_t)(void*)As;

  // ---- TDM: DMA the 16 x Kp A-tile (contiguous rows) into LDS, wave 0 issues ----
  if (wave == 0) {
    const unsigned long long ga = (unsigned long long)(const void*)(A + (size_t)m0 * Kp);
    // D# group 0: count=1, lds_addr, global_addr, type=2 ("image")
    v4u g0;
    g0[0] = 1u;                                   // count=1 valid descriptor
    g0[1] = lds_off;                              // lds_addr (bytes)
    g0[2] = (unsigned int)(ga & 0xFFFFFFFFull);   // global_addr[31:0]
    g0[3] = (unsigned int)((ga >> 32) & 0x1FFFFFFull) | (2u << 30); // addr[56:32] | type=2
    // D# group 1: data_size=1 (2B), tensor 2D Kp x 16, tile Kp x 16, contiguous stride
    v8i g1;
    g1[0] = (int)(1u << 16);                      // wg_mask=0, data_size=1 (2 bytes)
    g1[1] = (int)(((unsigned)Kp & 0xFFFFu) << 16);            // tensor_dim0[15:0] @63:48
    g1[2] = (int)((((unsigned)Kp >> 16) & 0xFFFFu) | (16u << 16)); // td0[31:16] | tensor_dim1
    g1[3] = (int)(((unsigned)Kp & 0xFFFFu) << 16);            // tile_dim0 @127:112
    g1[4] = 16;                                   // tile_dim1=16, tile_dim2=0
    g1[5] = Kp;                                   // tensor_dim0_stride[31:0]
    g1[6] = 0;                                    // stride hi / dim1_stride lo
    g1[7] = 0;
    v4i gz = {0, 0, 0, 0};                        // groups 2/3 unused (2D tensor)
#if defined(__clang_major__) && __clang_major__ >= 23
    v8i gz8 = {0, 0, 0, 0, 0, 0, 0, 0};
    __builtin_amdgcn_tensor_load_to_lds(g0, g1, gz, gz, gz8, 0);
#else
    __builtin_amdgcn_tensor_load_to_lds(g0, g1, gz, gz, 0);
#endif
    __builtin_amdgcn_s_wait_tensorcnt(0);
    asm volatile("" ::: "memory");   // TDM wrote LDS behind the compiler's back
  }
  __syncthreads();

  // A fragment addressing (16-bit A 16x32 layout), rows now in LDS:
  //   lanes 0-15 : M=lane,    K = k0+[0..7] and k0+[16..23]
  //   lanes16-31 : M=lane-16, K = k0+[8..15] and k0+[24..31]
  const int ak = (lane >> 4) * 8;
  const unsigned short* Arow = As + (size_t)(lane & 15) * Kp;

  // B fragment addressing (16-bit B 32x16 layout):
  //   lanes 0-15 : N=lane,    K = k0+[0..7] and k0+[16..23]
  //   lanes16-31 : N=lane-16, K = k0+[8..15] and k0+[24..31]
  int bn = n0 + (lane & 15);
  if (bn >= Np) bn = Np - 1;                // pad lanes read a valid row; masked at store
  const int bk = (lane >> 4) * 8;
  const unsigned short* __restrict__ Brow = B + (size_t)bn * Kp;

  FragC acc;
  acc.v = (v8f){0.f, 0.f, 0.f, 0.f, 0.f, 0.f, 0.f, 0.f};

#pragma unroll 2
  for (int k0 = 0; k0 < Kp; k0 += 32) {
    FragAB a, b;
    a.q[0] = *(const uint4*)(Arow + k0 + ak);        // ds_load_b128
    a.q[1] = *(const uint4*)(Arow + k0 + ak + 16);
    b.q[0] = *(const uint4*)(Brow + k0 + bk);        // global_load_b128
    b.q[1] = *(const uint4*)(Brow + k0 + bk + 16);
    __builtin_prefetch(Brow + k0 + 64, 0, 1);        // global_prefetch_b8
    acc.v = __builtin_amdgcn_wmma_f32_16x16x32_bf16(
        /*neg_a=*/false, a.v, /*neg_b=*/false, b.v,
        /*c_mod=*/(short)0, acc.v, /*reuse_a=*/false, /*reuse_b=*/false);
  }

  // C/D layout: VGPR i -> M = m0 + (lane<16 ? i : 8+i), N = n0 + (lane&15)
  const int cn = n0 + (lane & 15);
  if (cn < Nreal) {
    const float bv = bias ? bias[cn] : 0.0f;
    const int mhi = (lane >> 4) * 8;
#pragma unroll
    for (int i = 0; i < 8; ++i) {
      const int cm = m0 + mhi + i;
      float v = acc.f[i] + bv;
      if (relu) v = fmaxf(v, 0.0f);
      if (outF) outF[(size_t)cm * ldf + cn] = v;
      if (outB) outB[(size_t)cm * ldb + col_off + cn] = f2bf(v);
    }
  }
}

// ---------------- sequential LSTM recurrence, one block per direction ----------------
// xg: T x 1024 fp32 (precomputed x @ w_ih^T + b_ih + b_hh), whh: 1024 x 256 bf16
// writes bf16 h into lstm_out (T x 512): fwd -> cols [0,256), bwd -> cols [256,512)
__global__ __launch_bounds__(1024)
void lstm_kernel(const float* __restrict__ xg_f, const float* __restrict__ xg_b,
                 const unsigned short* __restrict__ whh_f,
                 const unsigned short* __restrict__ whh_b,
                 const float* __restrict__ h0, const float* __restrict__ c0,
                 unsigned short* __restrict__ lstm_out) {
  const int dir = blockIdx.x;
  const float* __restrict__ xg = dir ? xg_b : xg_f;
  const unsigned short* __restrict__ whh = dir ? whh_b : whh_f;

  __shared__ float hs[HH];
  __shared__ float cs[HH];
  __shared__ float gs[G4];

  const int tid = threadIdx.x;
  if (tid < HH) { hs[tid] = h0[dir * HH + tid]; cs[tid] = c0[dir * HH + tid]; }
  __syncthreads();

  const unsigned short* __restrict__ wrow = whh + (size_t)tid * HH;

  for (int t = 0; t < T_LEN; ++t) {
    const int row = dir ? (T_LEN - 1 - t) : t;
    float acc = xg[(size_t)row * G4 + tid];
#pragma unroll 8
    for (int k = 0; k < HH; k += 2) {
      unsigned int pk = *(const unsigned int*)(wrow + k);
      acc = fmaf(bf2f((unsigned short)pk),         hs[k],     acc);
      acc = fmaf(bf2f((unsigned short)(pk >> 16)), hs[k + 1], acc);
    }
    gs[tid] = acc;
    __syncthreads();
    if (tid < HH) {
      float ig = sigmoidf(gs[tid]);
      float fg = sigmoidf(gs[HH + tid]);
      float gg = tanhf(gs[2 * HH + tid]);
      float og = sigmoidf(gs[3 * HH + tid]);
      float c  = fg * cs[tid] + ig * gg;
      float h  = og * tanhf(c);
      cs[tid] = c;
      hs[tid] = h;
      lstm_out[(size_t)row * HID + dir * HH + tid] = f2bf(h);
    }
    __syncthreads();
  }
}

// ---------------- Viterbi forward scan ----------------
__global__ __launch_bounds__(128)
void viterbi_fwd_kernel(const float* __restrict__ feats,   // T x 16
                        const float* __restrict__ trans,   // 11 x 11
                        int* __restrict__ bps,             // T x 16
                        float* __restrict__ fv_out) {
  __shared__ float fv[16];
  __shared__ float tr[KTAGS * KTAGS];
  const int tid = threadIdx.x;
  if (tid < KTAGS * KTAGS) tr[tid] = trans[tid];
  if (tid < 16) fv[tid] = (tid == START_TAG) ? 0.0f : NEGV;
  __syncthreads();

  for (int t = 0; t < T_LEN; ++t) {
    float best = 0.0f; int bj = 0;
    if (tid < KTAGS) {
      best = fv[0] + tr[tid * KTAGS];
#pragma unroll
      for (int j = 1; j < KTAGS; ++j) {
        float s = fv[j] + tr[tid * KTAGS + j];
        if (s > best) { best = s; bj = j; }
      }
      best += feats[(size_t)t * 16 + tid];
    }
    __syncthreads();
    if (tid < KTAGS) { fv[tid] = best; bps[(size_t)t * 16 + tid] = bj; }
    __syncthreads();
  }
  if (tid < 16) fv_out[tid] = fv[tid];
}

// ---------------- terminal + backtrace (single thread) ----------------
__global__ void viterbi_back_kernel(const float* __restrict__ fv,
                                    const float* __restrict__ trans,
                                    const int* __restrict__ bps,
                                    float* __restrict__ score_out,
                                    int* __restrict__ path_out) {
  float best = fv[0] + trans[STOP_TAG * KTAGS];
  int bi = 0;
  for (int i = 1; i < KTAGS; ++i) {
    float s = fv[i] + trans[STOP_TAG * KTAGS + i];
    if (s > best) { best = s; bi = i; }
  }
  score_out[0] = best;
  int tag = bi;
  for (int t = T_LEN - 1; t >= 0; --t) {
    path_out[t] = tag;
    tag = bps[(size_t)t * 16 + tag];
  }
}

// ---------------- host-side launch ----------------
extern "C" void kernel_launch(void* const* d_in, const int* in_sizes, int n_in,
                              void* d_out, int out_size, void* d_ws, size_t ws_size,
                              hipStream_t stream) {
  (void)in_sizes; (void)n_in; (void)out_size; (void)ws_size;

  const float* s1    = (const float*)d_in[0];
  const float* s2    = (const float*)d_in[1];
  const float* dW    = (const float*)d_in[2];
  const float* db    = (const float*)d_in[3];
  const float* wihf  = (const float*)d_in[4];
  const float* whhf  = (const float*)d_in[5];
  const float* bihf  = (const float*)d_in[6];
  const float* bhhf  = (const float*)d_in[7];
  const float* wihb  = (const float*)d_in[8];
  const float* whhb  = (const float*)d_in[9];
  const float* bihb  = (const float*)d_in[10];
  const float* bhhb  = (const float*)d_in[11];
  const float* tagW  = (const float*)d_in[12];
  const float* tagb  = (const float*)d_in[13];
  const float* h0    = (const float*)d_in[14];
  const float* c0    = (const float*)d_in[15];
  const float* trans = (const float*)d_in[16];

  // carve workspace
  char* ws = (char*)d_ws;
  size_t off = 0;
  auto carve = [&](size_t bytes) -> void* {
    void* p = ws + off;
    off = (off + bytes + 255) & ~(size_t)255;
    return p;
  };
  unsigned short* s1b  = (unsigned short*)carve((size_t)T_LEN * D1P * 2);
  unsigned short* Xb   = (unsigned short*)carve((size_t)T_LEN * DINP * 2);
  unsigned short* dWb  = (unsigned short*)carve((size_t)DENSE_NP * D1P * 2);
  unsigned short* Wif  = (unsigned short*)carve((size_t)G4 * DINP * 2);
  unsigned short* Wib  = (unsigned short*)carve((size_t)G4 * DINP * 2);
  unsigned short* Whf  = (unsigned short*)carve((size_t)G4 * HH * 2);
  unsigned short* Whb  = (unsigned short*)carve((size_t)G4 * HH * 2);
  unsigned short* tWb  = (unsigned short*)carve((size_t)16 * HID * 2);
  float* biasf         = (float*)carve(G4 * 4);
  float* biasb         = (float*)carve(G4 * 4);
  float* xgf           = (float*)carve((size_t)T_LEN * G4 * 4);
  float* xgb           = (float*)carve((size_t)T_LEN * G4 * 4);
  unsigned short* lout = (unsigned short*)carve((size_t)T_LEN * HID * 2);
  float* feats         = (float*)carve((size_t)T_LEN * 16 * 4);
  int* bps             = (int*)carve((size_t)T_LEN * 16 * 4);
  float* fvbuf         = (float*)carve(64);

  const int TB = 256;
  auto blocks = [](long long n, int tb) { return (unsigned)((n + tb - 1) / tb); };

  // 1) bf16 conversions with padding
  cvt_pad_bf16<<<blocks((long long)T_LEN * D1P, TB), TB, 0, stream>>>(s1, s1b, T_LEN, D1, T_LEN, D1P);
  cvt_pad_bf16<<<blocks((long long)DENSE_NP * D1P, TB), TB, 0, stream>>>(dW, dWb, DENSE_N, D1, DENSE_NP, D1P);
  cvt_pad_bf16<<<blocks((long long)G4 * DINP, TB), TB, 0, stream>>>(wihf, Wif, G4, DIN, G4, DINP);
  cvt_pad_bf16<<<blocks((long long)G4 * DINP, TB), TB, 0, stream>>>(wihb, Wib, G4, DIN, G4, DINP);
  cvt_pad_bf16<<<blocks((long long)G4 * HH, TB), TB, 0, stream>>>(whhf, Whf, G4, HH, G4, HH);
  cvt_pad_bf16<<<blocks((long long)G4 * HH, TB), TB, 0, stream>>>(whhb, Whb, G4, HH, G4, HH);
  cvt_pad_bf16<<<blocks((long long)16 * HID, TB), TB, 0, stream>>>(tagW, tWb, KTAGS, HID, 16, HID);
  fill_x_kernel<<<blocks((long long)T_LEN * (DINP - DENSE_N), TB), TB, 0, stream>>>(s2, Xb);
  add_bias_kernel<<<blocks(G4, TB), TB, 0, stream>>>(bihf, bhhf, biasf, G4);
  add_bias_kernel<<<blocks(G4, TB), TB, 0, stream>>>(bihb, bhhb, biasb, G4);

  // 2) dense layer: relu(s1 @ dW^T + db) -> bf16 into Xb cols [0,200)
  {
    dim3 grid(T_LEN / 16, (DENSE_NP + 127) / 128);
    wmma_gemm_kernel<<<grid, 256, 0, stream>>>(s1b, dWb, db,
        /*outF*/nullptr, 0, /*outB*/Xb, DINP, 0,
        T_LEN, DENSE_N, DENSE_NP, D1P, /*relu*/1);
  }
  // 3) gate pre-activations: xg = Xb @ Wih^T + (b_ih + b_hh), both directions
  {
    dim3 grid(T_LEN / 16, G4 / 128);
    wmma_gemm_kernel<<<grid, 256, 0, stream>>>(Xb, Wif, biasf,
        xgf, G4, nullptr, 0, 0, T_LEN, G4, G4, DINP, 0);
    wmma_gemm_kernel<<<grid, 256, 0, stream>>>(Xb, Wib, biasb,
        xgb, G4, nullptr, 0, 0, T_LEN, G4, G4, DINP, 0);
  }
  // 4) sequential bidirectional LSTM recurrence (2 WGPs, one per direction)
  lstm_kernel<<<2, 1024, 0, stream>>>(xgf, xgb, Whf, Whb, h0, c0, lout);

  // 5) tag projection: feats = lstm_out @ tagW^T + tagb
  {
    dim3 grid(T_LEN / 16, 1);
    wmma_gemm_kernel<<<grid, 256, 0, stream>>>(lout, tWb, tagb,
        feats, 16, nullptr, 0, 0, T_LEN, KTAGS, 16, HID, 0);
  }
  // 6) Viterbi forward + backtrace
  viterbi_fwd_kernel<<<1, 128, 0, stream>>>(feats, trans, bps, fvbuf);
  viterbi_back_kernel<<<1, 1, 0, stream>>>(fvbuf, trans, bps,
                                           (float*)d_out, ((int*)d_out) + 1);
}
